// SemanticMemoryCore_83253646065897
// MI455X (gfx1250) — compile-verified, matrix operand order checked by reading, not run on hardware
//
#include <hip/hip_runtime.h>
#include <cstddef>

typedef __bf16 bf16;
typedef __attribute__((ext_vector_type(16))) __bf16 v16bf;
typedef __attribute__((ext_vector_type(8)))  __bf16 v8bf;
typedef __attribute__((ext_vector_type(8)))  float  v8f;

#define DEV static __device__ __forceinline__

namespace smc {

constexpr int B = 64, T = 512, F = 512, A = 8, U = 32, S = 256, L = 384, NS = 64;
constexpr float LOG_NS = 4.1588830833596715f;   // ln(64)
// K dims padded up to multiples of 32 for the WMMA k-loop
constexpr int K_OBS = 1056;  // 2F+A = 1032 -> 1056
constexpr int K_SIP = 576;   // F+A+U+3 = 555 -> 576
constexpr int K_LS1 = 320;   // S+U+A+3 = 299 -> 320
constexpr int K_LU  = 768;   // 2L

constexpr size_t SCAN_SMEM =
    (size_t)16 * 768 * 2 * 3 +                                       // xst, yA, yB (bf16)
    (size_t)(16*256 + 16*384 + 16*64 + 16*32 + 16*4 + 16) * 4;       // f32 state

DEV float sigf(float x) { return 1.0f / (1.0f + expf(-x)); }
DEV float geluf(float x) { return 0.5f * x * (1.0f + erff(x * 0.70710678118654752f)); }

DEV float wsum(float v) {
#pragma unroll
  for (int o = 16; o > 0; o >>= 1) v += __shfl_xor(v, o, 32);
  return v;
}
DEV float wmaxr(float v) {
#pragma unroll
  for (int o = 16; o > 0; o >>= 1) v = fmaxf(v, __shfl_xor(v, o, 32));
  return v;
}

// Wave index as an SGPR value: all tile-assignment control flow derived from it
// becomes scalar branches (no EXEC masking around v_wmma, which requires EXEC
// to be all-ones).
DEV int wave_id() { return __builtin_amdgcn_readfirstlane((int)(threadIdx.x >> 5)); }

// A-fragment (16x32 bf16, M x K) from LDS row-major activations [16][Kp].
// Per ISA layout: lanes 0-15 rows M=0..15 hold K = 0..7 (v0-3) and 16..23 (v4-7);
// lanes 16-31 hold K = 8..15 and 24..31. Two 16B LDS loads per fragment.
template <int Kp>
DEV v16bf load_afrag(const bf16* xs, int kb, int lane) {
  const int m = lane & 15, half = lane >> 4;
  const bf16* xr = xs + (size_t)m * Kp + kb * 32 + 8 * half;
  v8bf lo = *(const v8bf*)xr;
  v8bf hi = *(const v8bf*)(xr + 16);
  return __builtin_shufflevector(lo, hi, 0,1,2,3,4,5,6,7,8,9,10,11,12,13,14,15);
}

// Fully templated workgroup GEMM: y[16][Np] = x[16][Kp] @ Wfrag.
// Wfrag is pre-swizzled so fragment (kb, nt) is 512 contiguous bf16
// (32 lanes x 16 elems -> two global_load_b128 per lane).
// Tile assignment is split so the accumulation k-loops contain NO control flow:
//   - NFULL = ntiles/NWAVES tiles per wave, guard-free (A-fragment LDS load
//     amortized over NFULL v_wmma issues per k-block)
//   - R = ntiles%NWAVES extra tiles for waves < R, in a separate k-loop behind
//     one uniform scalar branch (wave is an SGPR value).
template <int Kp, int Np, int NWAVES>
DEV void wg_gemm(const bf16* __restrict__ xs, const bf16* __restrict__ wfrag,
                 bf16* __restrict__ y, int wave, int lane) {
  constexpr int ntiles = Np >> 4, kblocks = Kp >> 5;
  constexpr int NFULL = ntiles / NWAVES;
  constexpr int R = ntiles % NWAVES;
  const int n = lane & 15, mb = (lane >> 4) << 3;
  if constexpr (NFULL > 0) {
    v8f acc[NFULL];
#pragma unroll
    for (int q = 0; q < NFULL; ++q)
      acc[q] = (v8f){0.f, 0.f, 0.f, 0.f, 0.f, 0.f, 0.f, 0.f};
    for (int kb = 0; kb < kblocks; ++kb) {
      v16bf a = load_afrag<Kp>(xs, kb, lane);
#pragma unroll
      for (int q = 0; q < NFULL; ++q) {
        const int nt = wave + q * NWAVES;
        v16bf b = *(const v16bf*)(wfrag + (((size_t)(kb * ntiles + nt)) << 9) +
                                  (lane << 4));
        acc[q] = __builtin_amdgcn_wmma_f32_16x16x32_bf16(false, a, false, b, (short)0,
                                                         acc[q], false, false);
      }
    }
#pragma unroll
    for (int q = 0; q < NFULL; ++q) {
      const int nt = wave + q * NWAVES;
#pragma unroll
      for (int r = 0; r < 8; ++r)
        y[(size_t)(mb + r) * Np + nt * 16 + n] = (bf16)acc[q][r];
    }
  }
  if constexpr (R > 0) {
    if (wave < R) {   // single uniform scalar branch around a private k-loop
      const int nt = NFULL * NWAVES + wave;
      v8f acc = (v8f){0.f, 0.f, 0.f, 0.f, 0.f, 0.f, 0.f, 0.f};
      for (int kb = 0; kb < kblocks; ++kb) {
        v16bf a = load_afrag<Kp>(xs, kb, lane);
        v16bf b = *(const v16bf*)(wfrag + (((size_t)(kb * ntiles + nt)) << 9) +
                                  (lane << 4));
        acc = __builtin_amdgcn_wmma_f32_16x16x32_bf16(false, a, false, b, (short)0,
                                                      acc, false, false);
      }
#pragma unroll
      for (int r = 0; r < 8; ++r)
        y[(size_t)(mb + r) * Np + nt * 16 + n] = (bf16)acc[r];
    }
  }
}

// One wave handles one row: out = scale * gelu(LN(y + bias; g, beta))
DEV void row_ln_gelu(const bf16* ysrc, int N, const float* bias, const float* g,
                     const float* beta, float scale, bf16* dst, int row, int lane) {
  const int cnt = N >> 5;
  float vals[16];
  float s = 0.f, s2 = 0.f;
  for (int i = 0; i < cnt; ++i) {
    const int c = lane + 32 * i;
    float v = (float)ysrc[(size_t)row * N + c] + bias[c];
    vals[i] = v; s += v; s2 += v * v;
  }
  s = wsum(s); s2 = wsum(s2);
  const float mean = s / (float)N;
  const float inv = rsqrtf(s2 / (float)N - mean * mean + 1e-5f);
  for (int i = 0; i < cnt; ++i) {
    const int c = lane + 32 * i;
    float v = (vals[i] - mean) * inv * g[c] + beta[c];
    dst[(size_t)row * N + c] = (bf16)(geluf(v) * scale);
  }
}

// ---------------- weight pre-swizzle: f32 [K,N] -> bf16 WMMA B fragments ----------
__global__ __launch_bounds__(256) void wconv_kernel(const float* __restrict__ W,
                                                    bf16* __restrict__ dst,
                                                    int K, int N, int Kp, int Np) {
  const int ntiles = Np >> 4;
  size_t idx = (size_t)blockIdx.x * 256 + threadIdx.x;
  if (idx >= (size_t)Kp * Np) return;
  const int j = (int)(idx & 15);
  const int l = (int)((idx >> 4) & 31);
  const size_t tile = idx >> 9;
  const int nt = (int)(tile % ntiles);
  const int kb = (int)(tile / ntiles);
  const int half = l >> 4, v = j >> 1, p = j & 1;
  const int kl = ((v < 4) ? 0 : 16) + 8 * half + 2 * (v & 3) + p;
  const int k = kb * 32 + kl, n = nt * 16 + (l & 15);
  float val = (k < K && n < N) ? W[(size_t)k * N + n] : 0.f;
  dst[idx] = (bf16)val;
}

// ---------------- tiny projections: unc_raw (A->U, LN+GELU) and conf -------------
__global__ __launch_bounds__(256) void unc_conf_kernel(
    const float* __restrict__ aux, const float* __restrict__ uW,
    const float* __restrict__ ub, const float* __restrict__ ug,
    const float* __restrict__ ube, const float* __restrict__ cW,
    const float* __restrict__ cb, float* __restrict__ unc_raw,
    float* __restrict__ conf) {
  const int lane = threadIdx.x & 31;
  const size_t row = (size_t)blockIdx.x * 8 + wave_id();
  if (row >= (size_t)B * T) return;
  float a8[8];
#pragma unroll
  for (int i = 0; i < 8; ++i) a8[i] = aux[row * 8 + i];
  float v = ub[lane];
#pragma unroll
  for (int i = 0; i < 8; ++i) v += a8[i] * uW[i * 32 + lane];
  const float s = wsum(v), s2 = wsum(v * v);
  const float mean = s / 32.f;
  const float inv = rsqrtf(s2 / 32.f - mean * mean + 1e-5f);
  const float x = (v - mean) * inv * ug[lane] + ube[lane];
  unc_raw[row * 32 + lane] = geluf(x);
  if (lane == 0) {
    float c = cb[0];
#pragma unroll
    for (int i = 0; i < 8; ++i) c += a8[i] * cW[i];
    conf[row] = sigf(c);
  }
}

// ---------------- big GEMM: fused_obs = gelu(LN(concat @ obs_W + b)) -------------
__global__ __launch_bounds__(256) void obs_gemm_kernel(
    const float* __restrict__ sso, const float* __restrict__ eso,
    const float* __restrict__ aux, const bf16* __restrict__ wfrag,
    const float* __restrict__ ob, const float* __restrict__ og,
    const float* __restrict__ obe, float* __restrict__ out_fused) {
  __shared__ __align__(16) bf16 xs[16 * K_OBS];
  __shared__ __align__(16) bf16 ys[16 * F];
  const int tid = threadIdx.x, lane = tid & 31;
  const int wave = wave_id();
  const size_t row0 = (size_t)blockIdx.x * 16;   // rows in flattened [B*T]
  for (int i = tid; i < 16 * K_OBS; i += 256) {
    const int mm = i / K_OBS, k = i % K_OBS;
    const size_t r = row0 + mm;
    float v;
    if (k < F)              v = sso[r * F + k];
    else if (k < 2 * F)     v = eso[r * F + (k - F)];
    else if (k < 2 * F + A) v = aux[r * A + (k - 2 * F)];
    else                    v = 0.f;
    xs[i] = (bf16)v;
  }
  __syncthreads();
  wg_gemm<K_OBS, F, 8>(xs, wfrag, ys, wave, lane);
  __syncthreads();
  for (int mm = wave; mm < 16; mm += 8) {
    float vals[16];
    float s = 0.f, s2 = 0.f;
#pragma unroll
    for (int i = 0; i < 16; ++i) {
      const int c = lane + 32 * i;
      float v = (float)ys[mm * F + c] + ob[c];
      vals[i] = v; s += v; s2 += v * v;
    }
    s = wsum(s); s2 = wsum(s2);
    const float mean = s / (float)F;
    const float inv = rsqrtf(s2 / (float)F - mean * mean + 1e-5f);
#pragma unroll
    for (int i = 0; i < 16; ++i) {
      const int c = lane + 32 * i;
      float v = (vals[i] - mean) * inv * og[c] + obe[c];
      out_fused[(row0 + mm) * F + c] = geluf(v);
    }
  }
}

// ---------------- persistent scan kernel: 4 WGs x 16 batch rows ------------------
struct ScanParams {
  const float *aux, *post, *unc_raw, *conf, *fused;
  const bf16 *sipW, *wih, *whh, *ls1W, *ls2W, *gateW, *candW;
  const float *sip_b, *sip_g, *sip_be;
  const float *bih, *bhh;
  const float *ls_b1, *ls_g, *ls_be, *ls_b2;
  const float *gate_b, *cand_b;
  float *o_short, *o_long, *o_gate, *o_cov, *o_csum, *o_unc;
};

__global__ __launch_bounds__(512) void scan_kernel(ScanParams P) {
  extern __shared__ char smem_raw[];
  bf16* xst = (bf16*)smem_raw;          // [16][<=768] staging (WMMA A source)
  bf16* yA  = xst + 16 * 768;           // GEMM outputs
  bf16* yB  = yA + 16 * 768;
  float* h    = (float*)(yB + 16 * 768);  // sem_short [16][256]
  float* mL   = h + 16 * 256;             // sem_long  [16][384]
  float* cov  = mL + 16 * 384;            // coverage  [16][64]
  float* unc  = cov + 16 * 64;            // uncertainty [16][32]
  float* csum = unc + 16 * 32;            // cov summary [16][4]
  float* cf   = csum + 16 * 4;            // conf [16]
  const int tid = threadIdx.x, lane = tid & 31;
  const int wave = wave_id();             // SGPR wave index
  const int wg = blockIdx.x;              // 0..3 -> batch rows wg*16..+15

  for (int i = tid; i < 16 * 256; i += 512) h[i] = 0.f;
  for (int i = tid; i < 16 * 384; i += 512) mL[i] = 0.f;
  for (int i = tid; i < 16 * 64;  i += 512) cov[i] = 0.f;
  for (int i = tid; i < 16 * 32;  i += 512) unc[i] = 0.f;
  __syncthreads();

  for (int t = 0; t < T; ++t) {
    // ---- EMA updates (uncertainty U=32: wave=row, lane=u; coverage NS=64) ----
    {
      const int mm = wave, u = lane;
      const size_t b = (size_t)wg * 16 + mm;
      float v = 0.9f * unc[mm * 32 + u] + 0.1f * P.unc_raw[(b * T + t) * U + u];
      unc[mm * 32 + u] = v;
      P.o_unc[(b * T + t) * U + u] = v;
    }
    for (int i = tid; i < 16 * NS; i += 512) {
      const int mm = i >> 6, sx = i & 63;
      const size_t b = (size_t)wg * 16 + mm;
      float c = 0.95f * cov[i] + 0.05f * P.post[(b * T + t) * NS + sx];
      cov[i] = c;
      P.o_cov[(b * T + t) * NS + sx] = c;
    }
    if (tid < 16) cf[tid] = P.conf[((size_t)wg * 16 + tid) * T + t];
    __syncthreads();

    // ---- coverage summary [max, mean, norm-entropy] (one wave per row) ----
    {
      const int mm = wave;
      const float c0 = cov[mm * 64 + lane], c1 = cov[mm * 64 + 32 + lane];
      const float s = wsum(c0 + c1);
      const float mx = wmaxr(fmaxf(c0, c1));
      const float cs = fmaxf(s, 1e-8f);
      const float p0 = fmaxf(c0 / cs, 1e-8f), p1 = fmaxf(c1 / cs, 1e-8f);
      const float e = wsum(-(p0 * logf(p0) + p1 * logf(p1)));
      if (lane == 0) {
        csum[mm * 4 + 0] = mx; csum[mm * 4 + 1] = s / 64.f; csum[mm * 4 + 2] = e / LOG_NS;
        const size_t b = (size_t)wg * 16 + mm;
        P.o_csum[(b * T + t) * 3 + 0] = mx;
        P.o_csum[(b * T + t) * 3 + 1] = s / 64.f;
        P.o_csum[(b * T + t) * 3 + 2] = e / LOG_NS;
      }
    }
    __syncthreads();

    // ---- sip input [fused(512), aux(8), unc(32), csum(3), pad] -> [16][576] ----
    for (int i = tid; i < 16 * K_SIP; i += 512) {
      const int mm = i / K_SIP, k = i % K_SIP;
      const size_t b = (size_t)wg * 16 + mm;
      float v;
      if (k < F)                  v = P.fused[(b * T + t) * F + k];
      else if (k < F + A)         v = P.aux[(b * T + t) * A + (k - F)];
      else if (k < F + A + U)     v = unc[mm * 32 + (k - F - A)];
      else if (k < F + A + U + 3) v = csum[mm * 4 + (k - F - A - U)];
      else                        v = 0.f;
      xst[(size_t)mm * K_SIP + k] = (bf16)v;
    }
    if (t + 1 < T) {   // warm L2/WGP$ for next step's fused slab
      const size_t b = (size_t)wg * 16 + wave;
      __builtin_prefetch(&P.fused[(b * T + (t + 1)) * F + lane * 16], 0, 1);
    }
    __syncthreads();
    wg_gemm<K_SIP, S, 16>(xst, P.sipW, yA, wave, lane);
    __syncthreads();
    // short_in = conf * gelu(LN(y + sip_b)) -> xst [16][256]
    row_ln_gelu(yA, S, P.sip_b, P.sip_g, P.sip_be, cf[wave], xst, wave, lane);
    __syncthreads();
    wg_gemm<S, 3 * S, 16>(xst, P.wih, yA, wave, lane);     // gi
    __syncthreads();
    for (int i = tid; i < 16 * S; i += 512) xst[i] = (bf16)h[i];
    __syncthreads();
    wg_gemm<S, 3 * S, 16>(xst, P.whh, yB, wave, lane);     // gh
    __syncthreads();
    // ---- GRU combine (torch semantics) ----
    for (int i = tid; i < 16 * S; i += 512) {
      const int mm = i >> 8, j = i & 255;
      const float ir = (float)yA[mm * 768 + j]       + P.bih[j];
      const float iz = (float)yA[mm * 768 + 256 + j] + P.bih[256 + j];
      const float ii = (float)yA[mm * 768 + 512 + j] + P.bih[512 + j];
      const float hr = (float)yB[mm * 768 + j]       + P.bhh[j];
      const float hz = (float)yB[mm * 768 + 256 + j] + P.bhh[256 + j];
      const float hn = (float)yB[mm * 768 + 512 + j] + P.bhh[512 + j];
      const float r = sigf(ir + hr), z = sigf(iz + hz);
      const float nn = tanhf(ii + r * hn);
      const float hv = (1.f - z) * nn + z * h[i];
      h[i] = hv;
      P.o_short[(((size_t)wg * 16 + mm) * T + t) * S + j] = hv;
    }
    __syncthreads();
    // ---- ls1 input [h(256), unc(32), aux(8), csum(3), pad] -> [16][320] ----
    for (int i = tid; i < 16 * K_LS1; i += 512) {
      const int mm = i / K_LS1, k = i % K_LS1;
      float v;
      if (k < S)                  v = h[mm * 256 + k];
      else if (k < S + U)         v = unc[mm * 32 + (k - S)];
      else if (k < S + U + A)     v = P.aux[(((size_t)wg * 16 + mm) * T + t) * A + (k - S - U)];
      else if (k < S + U + A + 3) v = csum[mm * 4 + (k - S - U - A)];
      else                        v = 0.f;
      xst[(size_t)mm * K_LS1 + k] = (bf16)v;
    }
    __syncthreads();
    wg_gemm<K_LS1, L, 16>(xst, P.ls1W, yA, wave, lane);
    __syncthreads();
    row_ln_gelu(yA, L, P.ls_b1, P.ls_g, P.ls_be, 1.f, xst, wave, lane);
    __syncthreads();
    wg_gemm<L, L, 16>(xst, P.ls2W, yB, wave, lane);
    __syncthreads();
    // ---- gate/cand input [long_sum(384), sem_long(384)] -> [16][768] ----
    for (int i = tid; i < 16 * K_LU; i += 512) {
      const int mm = i / K_LU, k = i % K_LU;
      float v;
      if (k < L) v = (float)yB[mm * 384 + k] + P.ls_b2[k];
      else       v = mL[mm * 384 + (k - L)];
      xst[(size_t)mm * K_LU + k] = (bf16)v;
    }
    __syncthreads();
    wg_gemm<K_LU, L, 16>(xst, P.gateW, yA, wave, lane);
    wg_gemm<K_LU, L, 16>(xst, P.candW, yB, wave, lane);
    __syncthreads();
    // ---- slow gated long-memory update ----
    for (int i = tid; i < 16 * L; i += 512) {
      const int mm = i / L, j = i % L;
      const float g = sigf((float)yA[mm * 384 + j] + P.gate_b[j]) * 0.2f * cf[mm];
      const float c = tanhf((float)yB[mm * 384 + j] + P.cand_b[j]);
      const float mv = (1.f - g) * mL[i] + g * c;
      mL[i] = mv;
      const size_t b = (size_t)wg * 16 + mm;
      P.o_gate[(b * T + t) * L + j] = g;
      P.o_long[(b * T + t) * L + j] = mv;
    }
    __syncthreads();
  }
}

}  // namespace smc

extern "C" void kernel_launch(void* const* d_in, const int* in_sizes, int n_in,
                              void* d_out, int out_size, void* d_ws, size_t ws_size,
                              hipStream_t stream) {
  using namespace smc;
  (void)in_sizes; (void)n_in; (void)out_size; (void)ws_size;
  const float* sso    = (const float*)d_in[0];
  const float* eso    = (const float*)d_in[1];
  const float* aux    = (const float*)d_in[2];
  const float* post   = (const float*)d_in[3];
  const float* obs_W  = (const float*)d_in[4];
  const float* obs_b  = (const float*)d_in[5];
  const float* obs_g  = (const float*)d_in[6];
  const float* obs_be = (const float*)d_in[7];
  const float* sip_W  = (const float*)d_in[8];
  const float* sip_b  = (const float*)d_in[9];
  const float* sip_g  = (const float*)d_in[10];
  const float* sip_be = (const float*)d_in[11];
  const float* wih    = (const float*)d_in[12];
  const float* whh    = (const float*)d_in[13];
  const float* bih    = (const float*)d_in[14];
  const float* bhh    = (const float*)d_in[15];
  const float* unc_W  = (const float*)d_in[16];
  const float* unc_b  = (const float*)d_in[17];
  const float* unc_g  = (const float*)d_in[18];
  const float* unc_be = (const float*)d_in[19];
  const float* conf_W = (const float*)d_in[20];
  const float* conf_b = (const float*)d_in[21];
  const float* ls_W1  = (const float*)d_in[22];
  const float* ls_b1  = (const float*)d_in[23];
  const float* ls_g   = (const float*)d_in[24];
  const float* ls_be  = (const float*)d_in[25];
  const float* ls_W2  = (const float*)d_in[26];
  const float* ls_b2  = (const float*)d_in[27];
  const float* gate_W = (const float*)d_in[28];
  const float* gate_b = (const float*)d_in[29];
  const float* cand_W = (const float*)d_in[30];
  const float* cand_b = (const float*)d_in[31];

  float* out = (float*)d_out;
  float* o_short = out;
  float* o_long  = o_short + (size_t)B * T * S;
  float* o_gate  = o_long  + (size_t)B * T * L;
  float* o_cov   = o_gate  + (size_t)B * T * L;
  float* o_csum  = o_cov   + (size_t)B * T * NS;
  float* o_unc   = o_csum  + (size_t)B * T * 3;
  float* o_fused = o_unc   + (size_t)B * T * U;

  char* ws = (char*)d_ws;
  size_t off = 0;
  auto carve = [&](size_t bytes) -> char* {
    char* p = ws + off;
    off = (off + bytes + 255) & ~(size_t)255;
    return p;
  };
  bf16* f_obs  = (bf16*)carve((size_t)K_OBS * F * 2);
  bf16* f_sip  = (bf16*)carve((size_t)K_SIP * S * 2);
  bf16* f_wih  = (bf16*)carve((size_t)S * 3 * S * 2);
  bf16* f_whh  = (bf16*)carve((size_t)S * 3 * S * 2);
  bf16* f_ls1  = (bf16*)carve((size_t)K_LS1 * L * 2);
  bf16* f_ls2  = (bf16*)carve((size_t)L * L * 2);
  bf16* f_gate = (bf16*)carve((size_t)(2 * L) * L * 2);
  bf16* f_cand = (bf16*)carve((size_t)(2 * L) * L * 2);
  float* w_unc = (float*)carve((size_t)B * T * U * 4);
  float* w_cf  = (float*)carve((size_t)B * T * 4);

  auto wconv = [&](const float* W, bf16* dst, int K, int N, int Kp, int Np) {
    size_t total = (size_t)Kp * Np;
    int grid = (int)((total + 255) / 256);
    wconv_kernel<<<grid, 256, 0, stream>>>(W, dst, K, N, Kp, Np);
  };
  wconv(obs_W,  f_obs,  2 * F + A,     F,     K_OBS, F);
  wconv(sip_W,  f_sip,  F + A + U + 3, S,     K_SIP, S);
  wconv(wih,    f_wih,  S,             3 * S, S,     3 * S);
  wconv(whh,    f_whh,  S,             3 * S, S,     3 * S);
  wconv(ls_W1,  f_ls1,  S + U + A + 3, L,     K_LS1, L);
  wconv(ls_W2,  f_ls2,  L,             L,     L,     L);
  wconv(gate_W, f_gate, 2 * L,         L,     2 * L, L);
  wconv(cand_W, f_cand, 2 * L,         L,     2 * L, L);

  unc_conf_kernel<<<(B * T) / 8, 256, 0, stream>>>(aux, unc_W, unc_b, unc_g, unc_be,
                                                   conf_W, conf_b, w_unc, w_cf);

  obs_gemm_kernel<<<(B * T) / 16, 256, 0, stream>>>(sso, eso, aux, f_obs, obs_b,
                                                    obs_g, obs_be, o_fused);

  ScanParams P;
  P.aux = aux; P.post = post; P.unc_raw = w_unc; P.conf = w_cf; P.fused = o_fused;
  P.sipW = f_sip; P.wih = f_wih; P.whh = f_whh;
  P.ls1W = f_ls1; P.ls2W = f_ls2; P.gateW = f_gate; P.candW = f_cand;
  P.sip_b = sip_b; P.sip_g = sip_g; P.sip_be = sip_be;
  P.bih = bih; P.bhh = bhh;
  P.ls_b1 = ls_b1; P.ls_g = ls_g; P.ls_be = ls_be; P.ls_b2 = ls_b2;
  P.gate_b = gate_b; P.cand_b = cand_b;
  P.o_short = o_short; P.o_long = o_long; P.o_gate = o_gate;
  P.o_cov = o_cov; P.o_csum = o_csum; P.o_unc = o_unc;

  scan_kernel<<<B / 16, 512, (unsigned)SCAN_SMEM, stream>>>(P);
}
